// PosTrans_70489003262434
// MI455X (gfx1250) — matrix operand order repaired and checked by reading, compiled, not run on hardware
//
#include <hip/hip_runtime.h>
#include <hip/hip_bf16.h>
#include <stdint.h>

// ---------------------------------------------------------------------------
// GPT-style transformer forward (B=4,T=2048,E=768,HN=12,HS=64,L=6,FF=3072)
// for MI455X / gfx1250.
//  - All matmul operands staged as f16; WMMA v_wmma_f32_16x16x32_f16, f32 acc.
//  - Weights converted + transposed to WT[N,K] f16 once per use.
//  - Tile staging via Tensor Data Mover (tensor_load_to_lds + s_wait_tensorcnt),
//    double-buffered in the GEMM; vectorized b128 copy fallback.
// ---------------------------------------------------------------------------

#define B_ 4
#define T_ 2048
#define E_ 768
#define HN_ 12
#define HS_ 64
#define L_ 6
#define FF_ 3072
#define M_ (B_ * T_)   // 8192

typedef __attribute__((ext_vector_type(16))) _Float16 v16h;
typedef __attribute__((ext_vector_type(8)))  _Float16 v8h;
typedef __attribute__((ext_vector_type(4)))  _Float16 v4h;
typedef __attribute__((ext_vector_type(8)))  float    v8f;
typedef __attribute__((ext_vector_type(4)))  float    v4f;

static __device__ __forceinline__ v8f wmma_f16(v16h a, v16h b, v8f c) {
  return __builtin_amdgcn_wmma_f32_16x16x32_f16(
      false, a, false, b, (short)0, c, false, false);
}

// ---------------------------------------------------------------------------
// Tensor Data Mover 2D tile load (global f16 -> LDS) per CDNA5 D# spec.
// tensor dims == tile dims (no OOB), row stride in elements, optional LDS pad.
// ---------------------------------------------------------------------------
#if defined(__has_builtin)
#if __has_builtin(__builtin_amdgcn_tensor_load_to_lds) && \
    __has_builtin(__builtin_amdgcn_s_wait_tensorcnt)
#define PT_TDM 1
#endif
#endif

#if defined(PT_TDM)
typedef __attribute__((ext_vector_type(4))) unsigned int v4u_;
typedef __attribute__((ext_vector_type(8))) int          v8i_;
typedef __attribute__((ext_vector_type(4))) int          v4i_;

static __device__ __forceinline__ void pt_tdm_load2d(
    const void* lds, const void* gptr,
    unsigned tile_w /*elems*/, unsigned tile_h /*rows*/,
    unsigned row_stride /*elems*/,
    unsigned pad_interval_code, unsigned pad_amount_code) {
  const unsigned long long ga = (unsigned long long)(uintptr_t)gptr;
  const unsigned lab = (unsigned)(uintptr_t)lds;      // low 32 = LDS offset
  v4u_ g0;
  g0[0] = 1u;                                          // count=1, user mode
  g0[1] = lab;                                         // lds_addr
  g0[2] = (unsigned)ga;                                // global_addr[31:0]
  g0[3] = ((unsigned)(ga >> 32) & 0x01FFFFFFu) | (2u << 30);  // [56:32] | type=2
  v8i_ g1;
  g1[0] = (int)((1u << 16) |                           // data_size = 2 bytes
                (1u << 20) |                           // pad_enable
                (pad_interval_code << 22) | (pad_amount_code << 25));
  g1[1] = (int)((tile_w & 0xFFFFu) << 16);             // tensor_dim0 lo16
  g1[2] = (int)(((tile_w >> 16) & 0xFFFFu) | ((tile_h & 0xFFFFu) << 16));
  g1[3] = (int)(((tile_h >> 16) & 0xFFFFu) | ((tile_w & 0xFFFFu) << 16)); // tile_dim0
  g1[4] = (int)(tile_h & 0xFFFFu);                     // tile_dim1 (tile_dim2=0)
  g1[5] = (int)row_stride;                             // tensor_dim0_stride lo32
  g1[6] = 0;
  g1[7] = 0;
  v4i_ z4 = {0, 0, 0, 0};
#if __clang_major__ >= 23
  v8i_ z8 = {0, 0, 0, 0, 0, 0, 0, 0};
  __builtin_amdgcn_tensor_load_to_lds(g0, g1, z4, z4, z8, 0);
#else
  __builtin_amdgcn_tensor_load_to_lds(g0, g1, z4, z4, 0);
#endif
}
#endif  // PT_TDM

// ---------------------------------------------------------------------------
// LayerNorm over E=768 per row (f32 in, f16 out). One 256-thr block per row.
// ---------------------------------------------------------------------------
__global__ __launch_bounds__(256) void pt_layernorm(
    const float* __restrict__ x, const float* __restrict__ g,
    const float* __restrict__ bt, _Float16* __restrict__ out) {
  __shared__ float s1[256], s2[256];
  const int row = blockIdx.x, tid = threadIdx.x;
  const float* xr = x + (size_t)row * E_;
  float a0 = xr[tid], a1 = xr[tid + 256], a2 = xr[tid + 512];
  s1[tid] = a0 + a1 + a2;
  s2[tid] = a0 * a0 + a1 * a1 + a2 * a2;
  __syncthreads();
  for (int off = 128; off; off >>= 1) {
    if (tid < off) { s1[tid] += s1[tid + off]; s2[tid] += s2[tid + off]; }
    __syncthreads();
  }
  const float mean = s1[0] * (1.0f / E_);
  const float var  = s2[0] * (1.0f / E_) - mean * mean;
  const float rs   = rsqrtf(var + 1e-5f);
  _Float16* orow = out + (size_t)row * E_;
  orow[tid]       = (_Float16)((a0 - mean) * rs * g[tid]       + bt[tid]);
  orow[tid + 256] = (_Float16)((a1 - mean) * rs * g[tid + 256] + bt[tid + 256]);
  orow[tid + 512] = (_Float16)((a2 - mean) * rs * g[tid + 512] + bt[tid + 512]);
}

// ---------------------------------------------------------------------------
// Weight convert + transpose: W(f32, [K,N] or per-head [HN,K,64]) -> WT f16 [N,K]
// 32x32 tiles through LDS, vectorized loads/stores.
// ---------------------------------------------------------------------------
__global__ __launch_bounds__(256) void pt_wconv(
    const float* __restrict__ W, _Float16* __restrict__ WT,
    int K, int N, int headmode) {
  __shared__ float tile[32][33];
  const int n0 = blockIdx.x * 32, k0 = blockIdx.y * 32;
  const int tid = threadIdx.x;
  const int r = tid >> 3;          // 0..31
  const int c4 = (tid & 7) * 4;    // 0,4,..,28
  const float* src;
  if (headmode) {  // W[h][k][s], n = h*64 + s ; 32-wide tile stays in one head
    const int h = n0 >> 6, s0 = n0 & 63;
    src = W + ((size_t)h * K + (k0 + r)) * 64 + s0 + c4;
  } else {
    src = W + (size_t)(k0 + r) * N + n0 + c4;
  }
  const v4f tv = *(const v4f*)src;
#pragma unroll
  for (int j = 0; j < 4; ++j) tile[r][c4 + j] = tv[j];
  __syncthreads();
  v4h ov;
#pragma unroll
  for (int j = 0; j < 4; ++j) ov[j] = (_Float16)tile[c4 + j][r];
  *(v4h*)(WT + (size_t)(n0 + r) * K + k0 + c4) = ov;
}

// ---------------------------------------------------------------------------
// WMMA GEMM: C = op(A[M,K](f16) @ WT[N,K](f16)^T + bias) (+ residual)
// 64x64 tile / WG, 4 waves, K-step 32, TDM double-buffered staging.
// flags: 1 = exact GELU. Outputs: C32 (f32, with optional residual R) and/or C16.
// ---------------------------------------------------------------------------
#define GFLAG_GELU 1

static __device__ __forceinline__ void pt_stage_gemm(
    _Float16 (*__restrict__ Asb)[40], _Float16 (*__restrict__ Bsb)[40],
    const _Float16* __restrict__ A, const _Float16* __restrict__ WT,
    int m0, int n0, int K, int kt, int tid, int wave) {
#if defined(PT_TDM)
  if (wave == 0) {
    // 64 rows x 32 halves, row = 16 dwords (code 3), pad 4 dwords (code 3) -> 40
    pt_tdm_load2d(&Asb[0][0], A  + (size_t)m0 * K + kt * 32, 32, 64, K, 3, 3);
    pt_tdm_load2d(&Bsb[0][0], WT + (size_t)n0 * K + kt * 32, 32, 64, K, 3, 3);
  }
#else
  const int rr = tid >> 1, c0 = (tid & 1) * 16;
  const _Float16* pa = A  + (size_t)(m0 + rr) * K + kt * 32 + c0;
  const _Float16* pb = WT + (size_t)(n0 + rr) * K + kt * 32 + c0;
  v8h a0 = *(const v8h*)pa, a1 = *(const v8h*)(pa + 8);
  v8h b0 = *(const v8h*)pb, b1 = *(const v8h*)(pb + 8);
  *(v8h*)&Asb[rr][c0] = a0;  *(v8h*)&Asb[rr][c0 + 8] = a1;
  *(v8h*)&Bsb[rr][c0] = b0;  *(v8h*)&Bsb[rr][c0 + 8] = b1;
#endif
}

__global__ __launch_bounds__(128) void pt_gemm(
    const _Float16* __restrict__ A, const _Float16* __restrict__ WT,
    const float* __restrict__ bias, const float* __restrict__ R,
    float* __restrict__ C32, _Float16* __restrict__ C16,
    int M, int N, int K, int flags) {
  __shared__ __align__(16) _Float16 As[2][64][40];
  __shared__ __align__(16) _Float16 Bs[2][64][40];

  const int tid = threadIdx.x, lane = tid & 31, wave = tid >> 5;
  const int r = lane & 15, hi = lane >> 4;
  const int n0 = blockIdx.x * 64, m0 = blockIdx.y * 64;
  const int nK = K >> 5;

  v8f acc[4] = {};

  pt_stage_gemm(As[0], Bs[0], A, WT, m0, n0, K, 0, tid, wave);
#if defined(PT_TDM)
  if (wave == 0) __builtin_amdgcn_s_wait_tensorcnt(0);
#endif
  __syncthreads();

  for (int it = 0; it < nK; ++it) {
    const int cur = it & 1;
    if (it + 1 < nK)
      pt_stage_gemm(As[cur ^ 1], Bs[cur ^ 1], A, WT, m0, n0, K, it + 1, tid, wave);

    const _Float16 (*Ac)[40] = As[cur];
    const _Float16 (*Bc)[40] = Bs[cur];
    v8h a0 = *(const v8h*)&Ac[wave * 16 + r][hi * 8];
    v8h a1 = *(const v8h*)&Ac[wave * 16 + r][16 + hi * 8];
    v16h af = __builtin_shufflevector(a0, a1, 0, 1, 2, 3, 4, 5, 6, 7,
                                      8, 9, 10, 11, 12, 13, 14, 15);
#pragma unroll
    for (int nt = 0; nt < 4; ++nt) {
      v8h b0 = *(const v8h*)&Bc[nt * 16 + r][hi * 8];
      v8h b1 = *(const v8h*)&Bc[nt * 16 + r][16 + hi * 8];
      v16h bf = __builtin_shufflevector(b0, b1, 0, 1, 2, 3, 4, 5, 6, 7,
                                        8, 9, 10, 11, 12, 13, 14, 15);
      acc[nt] = wmma_f16(af, bf, acc[nt]);
    }

#if defined(PT_TDM)
    if (wave == 0) __builtin_amdgcn_s_wait_tensorcnt(0);
#endif
    __syncthreads();
  }

  const bool do_gelu = (flags & GFLAG_GELU) != 0;
#pragma unroll
  for (int nt = 0; nt < 4; ++nt) {
#pragma unroll
    for (int rr = 0; rr < 8; ++rr) {
      const int grow = m0 + wave * 16 + hi * 8 + rr;
      const int gcol = n0 + nt * 16 + r;
      float v = acc[nt][rr] + bias[gcol];
      if (R)       v += R[(size_t)grow * N + gcol];
      if (do_gelu) v = 0.5f * v * (1.0f + erff(v * 0.70710678118654752f));
      if (C32) C32[(size_t)grow * N + gcol] = v;
      if (C16) C16[(size_t)grow * N + gcol] = (_Float16)v;
    }
  }
}

// ---------------------------------------------------------------------------
// Flash attention, causal. q/k/v/o f16 [B,T,E], head h = cols h*64..h*64+63.
// WG = (b, head, 64 q rows); 4 waves x 16 rows; 32 keys per tile via TDM.
// scale = 1/sqrt(E) folded into q fragments.
// ---------------------------------------------------------------------------
__global__ __launch_bounds__(128) void pt_attn(
    const _Float16* __restrict__ Q, const _Float16* __restrict__ Kg,
    const _Float16* __restrict__ Vg, _Float16* __restrict__ O) {
  __shared__ __align__(16) _Float16 Kt[32][72];
  __shared__ __align__(16) _Float16 Vt[32][72];
  __shared__ __align__(16) _Float16 Pt[4][16][40];

  const int tid = threadIdx.x, lane = tid & 31, wave = tid >> 5;
  const int r = lane & 15, hi = lane >> 4;
  const int b = blockIdx.z, head = blockIdx.y;
  const int qrow0 = blockIdx.x * 64 + wave * 16;

  const _Float16* qp = Q  + (size_t)b * T_ * E_ + (size_t)head * HS_;
  const _Float16* kp = Kg + (size_t)b * T_ * E_ + (size_t)head * HS_;
  const _Float16* vp = Vg + (size_t)b * T_ * E_ + (size_t)head * HS_;
  _Float16*       op = O  + (size_t)b * T_ * E_ + (size_t)head * HS_;

  // q A-fragments for K-dim chunks [0,32) and [32,64), scaled by 1/sqrt(768)
  v16h aq0, aq1;
  {
    const _Float16 sc = (_Float16)0.03608439182435161f;
    const _Float16* qr = qp + (size_t)(qrow0 + r) * E_;
    v8h q0 = *(const v8h*)(qr + hi * 8)      * sc;
    v8h q1 = *(const v8h*)(qr + 16 + hi * 8) * sc;
    v8h q2 = *(const v8h*)(qr + 32 + hi * 8) * sc;
    v8h q3 = *(const v8h*)(qr + 48 + hi * 8) * sc;
    aq0 = __builtin_shufflevector(q0, q1, 0,1,2,3,4,5,6,7,8,9,10,11,12,13,14,15);
    aq1 = __builtin_shufflevector(q2, q3, 0,1,2,3,4,5,6,7,8,9,10,11,12,13,14,15);
  }

  v8f oa[4] = {};
  float m_i[8], l_i[8];
#pragma unroll
  for (int rr = 0; rr < 8; ++rr) { m_i[rr] = -3.0e38f; l_i[rr] = 0.0f; }

  const int nT = (blockIdx.x + 1) * 2;
  for (int jt = 0; jt < nT; ++jt) {
    const int j0 = jt * 32;
    __syncthreads();
#if defined(PT_TDM)
    if (wave == 0) {
      // 32 rows x 64 halves, row = 32 dwords (code 4), pad 4 dwords (code 3) -> 72
      pt_tdm_load2d(&Kt[0][0], kp + (size_t)j0 * E_, 64, 32, E_, 4, 3);
      pt_tdm_load2d(&Vt[0][0], vp + (size_t)j0 * E_, 64, 32, E_, 4, 3);
      __builtin_amdgcn_s_wait_tensorcnt(0);
    }
#else
    {
      const int kk = tid >> 2, c0 = (tid & 3) * 16;
      const _Float16* kr = kp + (size_t)(j0 + kk) * E_ + c0;
      const _Float16* vr = vp + (size_t)(j0 + kk) * E_ + c0;
      v8h k0v = *(const v8h*)kr, k1v = *(const v8h*)(kr + 8);
      v8h v0v = *(const v8h*)vr, v1v = *(const v8h*)(vr + 8);
      *(v8h*)&Kt[kk][c0] = k0v;  *(v8h*)&Kt[kk][c0 + 8] = k1v;
      *(v8h*)&Vt[kk][c0] = v0v;  *(v8h*)&Vt[kk][c0 + 8] = v1v;
    }
#endif
    __syncthreads();

    // S = q . k^T  (two 16-key subtiles, K=64 via two WMMAs each)
    v8f S0 = {}, S1 = {};
#pragma unroll
    for (int n = 0; n < 2; ++n) {
      v8f* Sp = n ? &S1 : &S0;
      v8h b0 = *(const v8h*)&Kt[n * 16 + r][hi * 8];
      v8h b1 = *(const v8h*)&Kt[n * 16 + r][16 + hi * 8];
      v16h bf = __builtin_shufflevector(b0, b1, 0,1,2,3,4,5,6,7,8,9,10,11,12,13,14,15);
      *Sp = wmma_f16(aq0, bf, *Sp);
      b0 = *(const v8h*)&Kt[n * 16 + r][32 + hi * 8];
      b1 = *(const v8h*)&Kt[n * 16 + r][32 + 16 + hi * 8];
      bf = __builtin_shufflevector(b0, b1, 0,1,2,3,4,5,6,7,8,9,10,11,12,13,14,15);
      *Sp = wmma_f16(aq1, bf, *Sp);
    }

    // online softmax (rows in VGPR idx rr; 16 cols across lanes of each half)
#pragma unroll
    for (int rr = 0; rr < 8; ++rr) {
      const int qi = qrow0 + hi * 8 + rr;
      float s0 = S0[rr]; if (j0 + r > qi)      s0 = -3.0e38f;
      float s1 = S1[rr]; if (j0 + 16 + r > qi) s1 = -3.0e38f;
      float rm = fmaxf(s0, s1);
#pragma unroll
      for (int off = 1; off < 16; off <<= 1)
        rm = fmaxf(rm, __shfl_xor(rm, off, 32));
      const float mnew = fmaxf(m_i[rr], rm);
      const float p0 = __expf(s0 - mnew);
      const float p1 = __expf(s1 - mnew);
      float ps = p0 + p1;
#pragma unroll
      for (int off = 1; off < 16; off <<= 1)
        ps += __shfl_xor(ps, off, 32);
      const float alpha = __expf(m_i[rr] - mnew);
      l_i[rr] = l_i[rr] * alpha + ps;
      m_i[rr] = mnew;
#pragma unroll
      for (int nt = 0; nt < 4; ++nt) oa[nt][rr] *= alpha;
      Pt[wave][hi * 8 + rr][r]      = (_Float16)p0;  // transpose via LDS
      Pt[wave][hi * 8 + rr][16 + r] = (_Float16)p1;
    }

    // O += P(16x32) @ V(32x64)
    v8h p0v = *(const v8h*)&Pt[wave][r][hi * 8];
    v8h p1v = *(const v8h*)&Pt[wave][r][16 + hi * 8];
    v16h ap = __builtin_shufflevector(p0v, p1v, 0,1,2,3,4,5,6,7,8,9,10,11,12,13,14,15);
#pragma unroll
    for (int nt = 0; nt < 4; ++nt) {
      v16h bf;
#pragma unroll
      for (int i = 0; i < 16; ++i) bf[i] = Vt[hi * 8 + i + (i >= 8 ? 8 : 0)][nt * 16 + r];
      oa[nt] = wmma_f16(ap, bf, oa[nt]);
    }
  }

#pragma unroll
  for (int nt = 0; nt < 4; ++nt)
#pragma unroll
    for (int rr = 0; rr < 8; ++rr) {
      const float outv = oa[nt][rr] / l_i[rr];
      op[(size_t)(qrow0 + hi * 8 + rr) * E_ + nt * 16 + r] = (_Float16)outv;
    }
}

// ---------------------------------------------------------------------------
// Final head: out[t] = act16[B-1, t, :] . hw + hb
// ---------------------------------------------------------------------------
__global__ __launch_bounds__(256) void pt_head(
    const _Float16* __restrict__ act, const float* __restrict__ hw,
    const float* __restrict__ hb, float* __restrict__ out) {
  __shared__ float red[256];
  const int t = blockIdx.x, tid = threadIdx.x;
  const _Float16* rowp = act + ((size_t)(B_ - 1) * T_ + t) * E_;
  red[tid] = (float)rowp[tid] * hw[tid] +
             (float)rowp[tid + 256] * hw[tid + 256] +
             (float)rowp[tid + 512] * hw[tid + 512];
  __syncthreads();
  for (int off = 128; off; off >>= 1) {
    if (tid < off) red[tid] += red[tid + off];
    __syncthreads();
  }
  if (tid == 0) out[t] = red[0] + hb[0];
}

// ---------------------------------------------------------------------------
extern "C" void kernel_launch(void* const* d_in, const int* in_sizes, int n_in,
                              void* d_out, int out_size, void* d_ws, size_t ws_size,
                              hipStream_t stream) {
  (void)in_sizes; (void)n_in; (void)out_size; (void)ws_size;
  const float* idx    = (const float*)d_in[0];
  const float* wq     = (const float*)d_in[2];
  const float* bq     = (const float*)d_in[3];
  const float* wk     = (const float*)d_in[4];
  const float* bk     = (const float*)d_in[5];
  const float* wv     = (const float*)d_in[6];
  const float* bv     = (const float*)d_in[7];
  const float* wproj  = (const float*)d_in[8];
  const float* bproj  = (const float*)d_in[9];
  const float* ln1_g  = (const float*)d_in[10];
  const float* ln1_b  = (const float*)d_in[11];
  const float* ln2_g  = (const float*)d_in[12];
  const float* ln2_b  = (const float*)d_in[13];
  const float* w1     = (const float*)d_in[14];
  const float* b1     = (const float*)d_in[15];
  const float* w2     = (const float*)d_in[16];
  const float* b2     = (const float*)d_in[17];
  const float* lnf_g  = (const float*)d_in[18];
  const float* lnf_b  = (const float*)d_in[19];
  const float* fw1    = (const float*)d_in[20];
  const float* fb1    = (const float*)d_in[21];
  const float* fw2    = (const float*)d_in[22];
  const float* fb2    = (const float*)d_in[23];
  const float* hw     = (const float*)d_in[24];
  const float* hb     = (const float*)d_in[25];
  float* out = (float*)d_out;

  const size_t ME = (size_t)M_ * E_;
  char* wsb = (char*)d_ws;
  float*    x   = (float*)wsb;                               // f32 residual
  _Float16* h16 = (_Float16*)(wsb + ME * 4);                 // LN out
  _Float16* q16 = (_Float16*)(wsb + ME * 4 + ME * 2);
  _Float16* k16 = q16 + ME;
  _Float16* v16 = k16 + ME;
  _Float16* o16 = v16 + ME;
  _Float16* ff16 = q16;            // M*FF halves == 4*ME halves: alias q..o
  _Float16* wt  = o16 + ME;        // transposed f16 weights, E*FF halves max

  hipMemcpyAsync(x, idx, ME * sizeof(float), hipMemcpyDeviceToDevice, stream);

  const dim3 gEE(E_ / 64, M_ / 64);
  const dim3 gEF(FF_ / 64, M_ / 64);
  const dim3 gAttn(T_ / 64, HN_, B_);
  const dim3 cEE(E_ / 32, E_ / 32);      // wconv grids: (N/32, K/32)
  const dim3 cEF(FF_ / 32, E_ / 32);
  const dim3 cFE(E_ / 32, FF_ / 32);

  for (int l = 0; l < L_; ++l) {
    pt_layernorm<<<M_, 256, 0, stream>>>(x, ln1_g + l * E_, ln1_b + l * E_, h16);

    pt_wconv<<<cEE, 256, 0, stream>>>(wq + (size_t)l * E_ * E_, wt, E_, E_, 1);
    pt_gemm<<<gEE, 128, 0, stream>>>(h16, wt, bq + l * E_, nullptr,
                                     nullptr, q16, M_, E_, E_, 0);
    pt_wconv<<<cEE, 256, 0, stream>>>(wk + (size_t)l * E_ * E_, wt, E_, E_, 1);
    pt_gemm<<<gEE, 128, 0, stream>>>(h16, wt, bk + l * E_, nullptr,
                                     nullptr, k16, M_, E_, E_, 0);
    pt_wconv<<<cEE, 256, 0, stream>>>(wv + (size_t)l * E_ * E_, wt, E_, E_, 1);
    pt_gemm<<<gEE, 128, 0, stream>>>(h16, wt, bv + l * E_, nullptr,
                                     nullptr, v16, M_, E_, E_, 0);

    pt_attn<<<gAttn, 128, 0, stream>>>(q16, k16, v16, o16);

    pt_wconv<<<cEE, 256, 0, stream>>>(wproj + (size_t)l * E_ * E_, wt, E_, E_, 0);
    pt_gemm<<<gEE, 128, 0, stream>>>(o16, wt, bproj + l * E_, x,
                                     x, nullptr, M_, E_, E_, 0);

    pt_layernorm<<<M_, 256, 0, stream>>>(x, ln2_g + l * E_, ln2_b + l * E_, h16);

    pt_wconv<<<cEF, 256, 0, stream>>>(w1 + (size_t)l * E_ * FF_, wt, E_, FF_, 0);
    pt_gemm<<<gEF, 128, 0, stream>>>(h16, wt, b1 + l * FF_, nullptr,
                                     nullptr, ff16, M_, FF_, E_, GFLAG_GELU);
    pt_wconv<<<cFE, 256, 0, stream>>>(w2 + (size_t)l * FF_ * E_, wt, FF_, E_, 0);
    pt_gemm<<<gEE, 128, 0, stream>>>(ff16, wt, b2 + l * E_, x,
                                     x, nullptr, M_, E_, FF_, 0);
  }

  pt_layernorm<<<M_, 256, 0, stream>>>(x, lnf_g, lnf_b, h16);
  pt_wconv<<<cEF, 256, 0, stream>>>(fw1, wt, E_, FF_, 0);
  pt_gemm<<<gEF, 128, 0, stream>>>(h16, wt, fb1, nullptr,
                                   nullptr, ff16, M_, FF_, E_, GFLAG_GELU);
  pt_wconv<<<cFE, 256, 0, stream>>>(fw2, wt, FF_, E_, 0);
  pt_gemm<<<gEE, 128, 0, stream>>>(ff16, wt, fb2, nullptr,
                                   nullptr, h16, M_, E_, FF_, 0);
  pt_head<<<T_, 256, 0, stream>>>(h16, hw, hb, out);
}
